// CompetitiveCrossAttentionBlock_68968584839947
// MI455X (gfx1250) — compile-verified, matrix operand order checked by reading, not run on hardware
//
#include <hip/hip_runtime.h>

// CompetitiveCrossAttentionBlock for MI455X (gfx1250), bf16 WMMA pipeline
// with Tensor-Data-Mover (TDM) staging of GEMM A-panels into LDS.
// B=4, S=2, T=1024, D=512, H=8, HD=64.

#define BB 4
#define SS 2
#define TT 1024
#define DD 512
#define HH 8
#define HD 64
#define BS (BB*SS)

typedef __attribute__((ext_vector_type(16))) __bf16 v16bf;
typedef __attribute__((ext_vector_type(8)))  __bf16 v8bf;
typedef __attribute__((ext_vector_type(4)))  __bf16 v4bf;
typedef __attribute__((ext_vector_type(8)))  float  v8f;
typedef __attribute__((ext_vector_type(4)))  unsigned int v4u;
typedef __attribute__((ext_vector_type(8)))  int    v8i;
typedef __attribute__((ext_vector_type(4)))  int    v4i;

__device__ __forceinline__ v16bf cat8(v8bf lo, v8bf hi) {
  return __builtin_shufflevector(lo, hi, 0,1,2,3,4,5,6,7,8,9,10,11,12,13,14,15);
}

#define WMMA_BF16(a, b, c) \
  __builtin_amdgcn_wmma_f32_16x16x32_bf16(false, (a), false, (b), (short)0, (c), false, false)

// ---------------------------------------------------------------------------
// TDM: 2-D tile load Global -> LDS via Tensor Data Mover (D# per ISA ch.8).
// dims/tiles in elements (data_size = 2 bytes, bf16). Issued by one wave,
// tracked with TENSORcnt.
// ---------------------------------------------------------------------------
__device__ __forceinline__ void tdm_load_2d(unsigned lds_addr, unsigned long long gaddr,
                                            unsigned dim0, unsigned dim1,
                                            unsigned tile0, unsigned tile1,
                                            unsigned long long stride0) {
  v4u g0;
  g0[0] = 1u;                                        // count=1 (valid user D#)
  g0[1] = lds_addr;                                  // lds_addr [63:32]
  g0[2] = (unsigned)(gaddr & 0xffffffffu);           // global_addr[31:0]
  g0[3] = (unsigned)((gaddr >> 32) & 0x01ffffffu)    // global_addr[56:32]
          | (2u << 30);                              // type=2 ("image")
  v8i g1;
  g1[0] = (int)(1u << 16);                           // data_size=1 (2B), mask=0
  g1[1] = (int)((dim0 & 0xffffu) << 16);             // tensor_dim0[15:0]
  g1[2] = (int)((dim0 >> 16) | ((dim1 & 0xffffu) << 16)); // dim0[31:16] | dim1[15:0]
  g1[3] = (int)((dim1 >> 16) | (tile0 << 16));       // dim1[31:16] | tile_dim0
  g1[4] = (int)(tile1 & 0xffffu);                    // tile_dim1 (tile_dim2=0)
  g1[5] = (int)(unsigned)(stride0 & 0xffffffffu);    // tensor_dim0_stride[31:0]
  g1[6] = (int)(unsigned)((stride0 >> 32) & 0xffffu);// stride0[47:32] (dim1_stride=0)
  g1[7] = 0;
  v4i z4 = 0;
#if __clang_major__ >= 23
  v8i z8 = 0;
  __builtin_amdgcn_tensor_load_to_lds(g0, g1, z4, z4, z8, 0);
#else
  __builtin_amdgcn_tensor_load_to_lds(g0, g1, z4, z4, 0);
#endif
}

// ---------------------------------------------------------------------------
// Kernel 1: f32 -> bf16 conversion (hidden states + weights)
// ---------------------------------------------------------------------------
__global__ __launch_bounds__(256) void k_cvt_bf16(const float* __restrict__ src,
                                                  __bf16* __restrict__ dst, int n) {
  int i = (blockIdx.x * 256 + threadIdx.x) * 4;
  if (i < n) {
    float4 v = *(const float4*)(src + i);
    v4bf o;
    o[0] = (__bf16)v.x; o[1] = (__bf16)v.y; o[2] = (__bf16)v.z; o[3] = (__bf16)v.w;
    *(v4bf*)(dst + i) = o;
  }
}

// ---------------------------------------------------------------------------
// Kernel 2: projection GEMM  out(t,e) = sum_d X[t,d] * W[e,d] + bias[e]
// Block tile 128(M) x 128(N), 8 waves (2M x 4N). The whole 128x512 bf16
// A-panel (128KB) is TDM-staged into LDS once; the K-loop runs barrier-free.
// mode 0: store [bs][h][t][hd] (Q,K).  mode 1: store [bs][h][hd][t] (V^T).
// ---------------------------------------------------------------------------
__global__ __launch_bounds__(256) void k_proj(const __bf16* __restrict__ X,
                                              const __bf16* __restrict__ W,
                                              const float* __restrict__ bias,
                                              __bf16* __restrict__ dst, int mode) {
  __shared__ __align__(16) __bf16 xt[128][512];   // 128KB A-panel
  const int tid = threadIdx.x;
  const int lane = tid & 31, wave = tid >> 5;
  const int wm = wave >> 2, wn = wave & 3;
  const int half = lane >> 4, nl = lane & 15;
  const int bs = blockIdx.z;
  const int m0 = blockIdx.x * 128;
  const int n0 = blockIdx.y * 128 + wn * 32;
  const __bf16* Xb = X + (size_t)bs * TT * DD;

  if (tid < 32) {   // wave 0 issues the DMA; TDM ignores EXEC, scalar-operand op
    unsigned lds_a = (unsigned)(size_t)&xt[0][0];
    unsigned long long ga = (unsigned long long)(size_t)(Xb + (size_t)m0 * DD);
    tdm_load_2d(lds_a, ga, DD, TT, DD, 128, DD);
    __builtin_amdgcn_s_wait_tensorcnt(0);
  }
  __syncthreads();

  v8f acc[4][2] = {};
  for (int ks = 0; ks < DD / 32; ++ks) {
    const int k0 = ks * 32;
    v16bf a[4], bfr[2];
    #pragma unroll
    for (int i = 0; i < 4; ++i) {
      int r = wm * 64 + i * 16 + nl;
      a[i] = cat8(*(const v8bf*)&xt[r][k0 + half * 8],
                  *(const v8bf*)&xt[r][k0 + 16 + half * 8]);
    }
    #pragma unroll
    for (int j = 0; j < 2; ++j) {
      const __bf16* wp = W + (size_t)(n0 + j * 16 + nl) * DD + k0;
      bfr[j] = cat8(*(const v8bf*)(wp + half * 8), *(const v8bf*)(wp + 16 + half * 8));
    }
    #pragma unroll
    for (int i = 0; i < 4; ++i)
      #pragma unroll
      for (int j = 0; j < 2; ++j)
        acc[i][j] = WMMA_BF16(a[i], bfr[j], acc[i][j]);
  }
  // epilogue: bias add, bf16 convert, scattered store in head layout
  #pragma unroll
  for (int i = 0; i < 4; ++i) {
    #pragma unroll
    for (int j = 0; j < 2; ++j) {
      int e = n0 + j * 16 + nl;
      float bv = bias[e];
      int h = e >> 6, hd = e & 63;
      #pragma unroll
      for (int jj = 0; jj < 8; ++jj) {
        int m = m0 + wm * 64 + i * 16 + half * 8 + jj;
        float v = acc[i][j][jj] + bv;
        size_t off = (mode == 0)
            ? (((size_t)bs * HH + h) * TT + m) * HD + hd
            : (((size_t)bs * HH + h) * HD + hd) * TT + m;
        dst[off] = (__bf16)v;
      }
    }
  }
}

// ---------------------------------------------------------------------------
// Kernel 3: fused competitive cross-attention for one (b, h, 16-row q block).
// Both streams processed together. Logits kept in LDS (flash-style, never
// touch HBM). 8 waves each own 128 of the 1024 k columns.
// ---------------------------------------------------------------------------
__global__ __launch_bounds__(256) void k_attn(const __bf16* __restrict__ Q,
                                              const __bf16* __restrict__ Kc,
                                              const __bf16* __restrict__ Vt,
                                              __bf16* __restrict__ Hout) {
  __shared__ __align__(16) unsigned char lraw[2 * 16 * 1024 * 4]; // 128KB logits
  __shared__ float lds_hacc[2][16][64];                            // 8KB
  __shared__ float lds_red[2][16][16];                             // 2KB
  float* l12 = (float*)lraw;                 // [16][1024]
  float* l21 = (float*)(lraw + 65536);       // [16][1024]
  __bf16* a12s = (__bf16*)lraw;              // [16][1024] (aliased after reads done)
  __bf16* a21s = (__bf16*)(lraw + 32768);    // [16][1024]

  const int tid = threadIdx.x, lane = tid & 31, wave = tid >> 5;
  const int half = lane >> 4, nl = lane & 15;
  const int q0 = blockIdx.x * 16, h = blockIdx.y, b = blockIdx.z;

  #pragma unroll
  for (int i = 0; i < 8; ++i) ((float*)lds_hacc)[tid * 8 + i] = 0.f;

  const size_t hb0 = ((size_t)(b * 2 + 0) * HH + h) * TT * HD;
  const size_t hb1 = ((size_t)(b * 2 + 1) * HH + h) * TT * HD;

  // Q fragments (16x32, two kd steps over HD=64), resident for whole phase
  v16bf q1[2], q2[2];
  #pragma unroll
  for (int kd = 0; kd < 2; ++kd) {
    const __bf16* p1 = Q + hb0 + (size_t)(q0 + nl) * HD + kd * 32;
    const __bf16* p2 = Q + hb1 + (size_t)(q0 + nl) * HD + kd * 32;
    q1[kd] = cat8(*(const v8bf*)(p1 + half * 8), *(const v8bf*)(p1 + 16 + half * 8));
    q2[kd] = cat8(*(const v8bf*)(p2 + half * 8), *(const v8bf*)(p2 + 16 + half * 8));
  }
  const float scale = 0.125f; // 1/sqrt(64)

  // ---- logits: each wave does 8 16-col tiles, 4 WMMAs per tile ----
  #pragma unroll
  for (int kt = 0; kt < 8; ++kt) {
    int kg = wave * 128 + kt * 16;
    const __bf16* pk2 = Kc + hb1 + (size_t)(kg + nl) * HD;
    const __bf16* pk1 = Kc + hb0 + (size_t)(kg + nl) * HD;
    v16bf k2f0 = cat8(*(const v8bf*)(pk2 + half * 8), *(const v8bf*)(pk2 + 16 + half * 8));
    v16bf k2f1 = cat8(*(const v8bf*)(pk2 + 32 + half * 8), *(const v8bf*)(pk2 + 48 + half * 8));
    v16bf k1f0 = cat8(*(const v8bf*)(pk1 + half * 8), *(const v8bf*)(pk1 + 16 + half * 8));
    v16bf k1f1 = cat8(*(const v8bf*)(pk1 + 32 + half * 8), *(const v8bf*)(pk1 + 48 + half * 8));
    v8f a12 = {}, a21 = {};
    a12 = WMMA_BF16(q1[0], k2f0, a12);
    a12 = WMMA_BF16(q1[1], k2f1, a12);
    a21 = WMMA_BF16(q2[0], k1f0, a21);
    a21 = WMMA_BF16(q2[1], k1f1, a21);
    #pragma unroll
    for (int jj = 0; jj < 8; ++jj) {
      int m = half * 8 + jj;
      l12[m * 1024 + kg + nl] = a12[jj] * scale;
      l21[m * 1024 + kg + nl] = a21[jj] * scale;
    }
  }
  __syncthreads();

  // ---- coupled softmax: 16 threads per row, 64 cols each ----
  const int r = tid >> 4, seg = tid & 15, c0 = seg * 64;
  float m12 = -3.0e38f, m21 = -3.0e38f;
  #pragma unroll 8
  for (int i = 0; i < 64; ++i) {
    m12 = fmaxf(m12, l12[r * 1024 + c0 + i]);
    m21 = fmaxf(m21, l21[r * 1024 + c0 + i]);
  }
  lds_red[0][r][seg] = m12;
  lds_red[1][r][seg] = m21;
  __syncthreads();
  m12 = -3.0e38f; m21 = -3.0e38f;
  #pragma unroll
  for (int i = 0; i < 16; ++i) {
    m12 = fmaxf(m12, lds_red[0][r][i]);
    m21 = fmaxf(m21, lds_red[1][r][i]);
  }
  __syncthreads();
  float e12c[64];
  float s12 = 0.f, s21 = 0.f;
  #pragma unroll
  for (int i = 0; i < 64; ++i) {
    float e = __expf(l12[r * 1024 + c0 + i] - m12);
    e12c[i] = e;
    s12 += e;
    s21 += __expf(l21[r * 1024 + c0 + i] - m21);
  }
  lds_red[0][r][seg] = s12;
  lds_red[1][r][seg] = s21;
  __syncthreads();
  s12 = 0.f; s21 = 0.f;
  #pragma unroll
  for (int i = 0; i < 16; ++i) { s12 += lds_red[0][r][i]; s21 += lds_red[1][r][i]; }
  const float inv12 = 1.f / s12, inv21 = 1.f / s21;
  // all reads of l12 region are complete for every thread; safe to alias-write
  #pragma unroll
  for (int i = 0; i < 64; ++i) {
    float S12 = e12c[i] * inv12;
    float S21 = __expf(l21[r * 1024 + c0 + i] - m21) * inv21;
    float Mv = S12 + S21 + 1e-6f;
    a12s[r * 1024 + c0 + i] = (__bf16)(S12 / Mv);
    a21s[r * 1024 + c0 + i] = (__bf16)(S21 / Mv);
  }
  __syncthreads();

  // ---- A @ V : per wave, 4 k-steps of 32 over its 128 cols, N=64 (4 tiles) ----
  v8f h1a[4] = {}, h2a[4] = {};
  const size_t vb1 = ((size_t)(b * 2 + 0) * HH + h) * HD * TT;
  const size_t vb2 = ((size_t)(b * 2 + 1) * HH + h) * HD * TT;
  #pragma unroll
  for (int kt = 0; kt < 4; ++kt) {
    int k0 = wave * 128 + kt * 32;
    const __bf16* pa12 = a12s + nl * 1024 + k0;
    const __bf16* pa21 = a21s + nl * 1024 + k0;
    v16bf af12 = cat8(*(const v8bf*)(pa12 + half * 8), *(const v8bf*)(pa12 + 16 + half * 8));
    v16bf af21 = cat8(*(const v8bf*)(pa21 + half * 8), *(const v8bf*)(pa21 + 16 + half * 8));
    #pragma unroll
    for (int nt = 0; nt < 4; ++nt) {
      const __bf16* pv2 = Vt + vb2 + (size_t)(nt * 16 + nl) * TT + k0;
      const __bf16* pv1 = Vt + vb1 + (size_t)(nt * 16 + nl) * TT + k0;
      v16bf v2f = cat8(*(const v8bf*)(pv2 + half * 8), *(const v8bf*)(pv2 + 16 + half * 8));
      v16bf v1f = cat8(*(const v8bf*)(pv1 + half * 8), *(const v8bf*)(pv1 + 16 + half * 8));
      h1a[nt] = WMMA_BF16(af12, v2f, h1a[nt]);
      h2a[nt] = WMMA_BF16(af21, v1f, h2a[nt]);
    }
  }
  // cross-wave reduction of partial products via LDS float atomics
  #pragma unroll
  for (int nt = 0; nt < 4; ++nt)
    #pragma unroll
    for (int jj = 0; jj < 8; ++jj) {
      int m = half * 8 + jj, n = nt * 16 + nl;
      atomicAdd(&lds_hacc[0][m][n], h1a[nt][jj]);
      atomicAdd(&lds_hacc[1][m][n], h2a[nt][jj]);
    }
  __syncthreads();
  // write H block (bf16) back to [b][s][t][d] with heads reassembled
  #pragma unroll
  for (int i = 0; i < 8; ++i) {
    int idx = tid * 8 + i;
    int st = idx >> 10, rem = idx & 1023;
    int rr = rem >> 6, cc = rem & 63;
    size_t off = (((size_t)(b * 2 + st) * TT) + q0 + rr) * DD + h * 64 + cc;
    Hout[off] = (__bf16)lds_hacc[st][rr][cc];
  }
}

// ---------------------------------------------------------------------------
// Kernel 4: output projection + bias + LayerNorm + gated residual.
// Block = 32 full rows; the 32x512 bf16 H panel is TDM-staged into LDS so
// all 8 waves share one copy instead of 8 redundant global streams.
// ---------------------------------------------------------------------------
__global__ __launch_bounds__(256) void k_out(const __bf16* __restrict__ Hb,
                                             const __bf16* __restrict__ Wo,
                                             const float* __restrict__ bo,
                                             const float* __restrict__ ln_g,
                                             const float* __restrict__ ln_b,
                                             const float* __restrict__ alpha,
                                             const float* __restrict__ hidden,
                                             float* __restrict__ out) {
  __shared__ __align__(16) __bf16 ht[32][512];  // 32KB TDM-staged A panel
  __shared__ float ybuf[32][512];               // 64KB
  __shared__ float rsum[32][8], rsq[32][8];
  const int tid = threadIdx.x, lane = tid & 31, wave = tid >> 5;
  const int half = lane >> 4, nl = lane & 15;
  const int m0 = blockIdx.x * 32, bs = blockIdx.y, s = bs & 1;
  const int n0 = wave * 64;

  if (tid < 32) {
    unsigned lds_a = (unsigned)(size_t)&ht[0][0];
    unsigned long long ga =
        (unsigned long long)(size_t)(Hb + (size_t)bs * TT * DD + (size_t)m0 * DD);
    tdm_load_2d(lds_a, ga, DD, TT, DD, 32, DD);
    __builtin_amdgcn_s_wait_tensorcnt(0);
  }
  __syncthreads();

  v8f acc[2][4] = {};
  for (int ks = 0; ks < DD / 32; ++ks) {
    const int k0 = ks * 32;
    v16bf a[2], bfr[4];
    #pragma unroll
    for (int i = 0; i < 2; ++i) {
      int r = i * 16 + nl;
      a[i] = cat8(*(const v8bf*)&ht[r][k0 + half * 8],
                  *(const v8bf*)&ht[r][k0 + 16 + half * 8]);
    }
    #pragma unroll
    for (int j = 0; j < 4; ++j) {
      const __bf16* p = Wo + (size_t)(n0 + j * 16 + nl) * DD + k0;
      bfr[j] = cat8(*(const v8bf*)(p + half * 8), *(const v8bf*)(p + 16 + half * 8));
    }
    #pragma unroll
    for (int i = 0; i < 2; ++i)
      #pragma unroll
      for (int j = 0; j < 4; ++j)
        acc[i][j] = WMMA_BF16(a[i], bfr[j], acc[i][j]);
  }
  #pragma unroll
  for (int i = 0; i < 2; ++i)
    #pragma unroll
    for (int j = 0; j < 4; ++j) {
      int n = n0 + j * 16 + nl;
      float bv = bo[n];
      #pragma unroll
      for (int jj = 0; jj < 8; ++jj)
        ybuf[i * 16 + half * 8 + jj][n] = acc[i][j][jj] + bv;
    }
  __syncthreads();

  // LayerNorm over 512 cols: 8 threads per row, 64 cols each
  const int r = tid >> 3, seg = tid & 7, c0 = seg * 64;
  float sm = 0.f, sq = 0.f;
  #pragma unroll 8
  for (int i = 0; i < 64; ++i) {
    float x = ybuf[r][c0 + i];
    sm += x; sq += x * x;
  }
  rsum[r][seg] = sm; rsq[r][seg] = sq;
  __syncthreads();
  sm = 0.f; sq = 0.f;
  #pragma unroll
  for (int i = 0; i < 8; ++i) { sm += rsum[r][i]; sq += rsq[r][i]; }
  const float mu = sm * (1.f / 512.f);
  const float var = sq * (1.f / 512.f) - mu * mu;
  const float rs = rsqrtf(var + 1e-5f);
  const float al = alpha[s];
  const size_t rowoff = ((size_t)bs * TT + m0 + r) * DD;
  #pragma unroll 8
  for (int i = 0; i < 64; ++i) {
    int e = c0 + i;
    float y = (ybuf[r][e] - mu) * rs * ln_g[s * DD + e] + ln_b[s * DD + e];
    out[rowoff + e] = hidden[rowoff + e] + y * al;
  }
}

// ---------------------------------------------------------------------------
// Launch
// ---------------------------------------------------------------------------
extern "C" void kernel_launch(void* const* d_in, const int* in_sizes, int n_in,
                              void* d_out, int out_size, void* d_ws, size_t ws_size,
                              hipStream_t stream) {
  const float* hs   = (const float*)d_in[0];
  const float* Wq   = (const float*)d_in[1];
  const float* bq   = (const float*)d_in[2];
  const float* Wk   = (const float*)d_in[3];
  const float* bk   = (const float*)d_in[4];
  const float* Wv   = (const float*)d_in[5];
  const float* bv   = (const float*)d_in[6];
  const float* Wo   = (const float*)d_in[7];
  const float* bo   = (const float*)d_in[8];
  const float* ln_g = (const float*)d_in[9];
  const float* ln_b = (const float*)d_in[10];
  const float* alph = (const float*)d_in[11];
  float* out = (float*)d_out;

  char* ws = (char*)d_ws;
  const size_t SZ_ACT = (size_t)BS * TT * DD * 2;   // 8 MiB per bf16 tensor
  const size_t SZ_W   = (size_t)DD * DD * 2;        // 512 KiB per bf16 weight
  __bf16* Xb  = (__bf16*)(ws);
  __bf16* Qb  = (__bf16*)(ws + 1 * SZ_ACT);
  __bf16* Kb  = (__bf16*)(ws + 2 * SZ_ACT);
  __bf16* Vtb = (__bf16*)(ws + 3 * SZ_ACT);
  __bf16* Hb  = (__bf16*)(ws + 4 * SZ_ACT);
  __bf16* Wqb = (__bf16*)(ws + 5 * SZ_ACT);
  __bf16* Wkb = (__bf16*)(ws + 5 * SZ_ACT + 1 * SZ_W);
  __bf16* Wvb = (__bf16*)(ws + 5 * SZ_ACT + 2 * SZ_W);
  __bf16* Wob = (__bf16*)(ws + 5 * SZ_ACT + 3 * SZ_W);

  const int n_act = BS * TT * DD;   // 4,194,304
  const int n_w   = DD * DD;        // 262,144
  k_cvt_bf16<<<n_act / 1024, 256, 0, stream>>>(hs, Xb, n_act);
  k_cvt_bf16<<<n_w / 1024, 256, 0, stream>>>(Wq, Wqb, n_w);
  k_cvt_bf16<<<n_w / 1024, 256, 0, stream>>>(Wk, Wkb, n_w);
  k_cvt_bf16<<<n_w / 1024, 256, 0, stream>>>(Wv, Wvb, n_w);
  k_cvt_bf16<<<n_w / 1024, 256, 0, stream>>>(Wo, Wob, n_w);

  dim3 gp(TT / 128, DD / 128, BS);
  k_proj<<<gp, 256, 0, stream>>>(Xb, Wqb, bq, Qb, 0);
  k_proj<<<gp, 256, 0, stream>>>(Xb, Wkb, bk, Kb, 0);
  k_proj<<<gp, 256, 0, stream>>>(Xb, Wvb, bv, Vtb, 1);

  k_attn<<<dim3(TT / 16, HH, BB), 256, 0, stream>>>(Qb, Kb, Vtb, Hb);

  k_out<<<dim3(TT / 32, BS), 256, 0, stream>>>(Hb, Wob, bo, ln_g, ln_b, alph, hs, out);
}
